// TemporalTransformer_71201967833645
// MI455X (gfx1250) — matrix operand order repaired and checked by reading, not compiled, not run on hardware
//
#include <hip/hip_runtime.h>

// ---------------------------------------------------------------------------
// MHA on gfx1250: f16 WMMA (v_wmma_f32_16x16x32_f16) for all four GEMMs,
// flash-style online softmax, TDM (tensor_load_to_lds) double-buffered K/V
// tile staging, LDS re-fragmentation of P, f32 accumulation.
// ---------------------------------------------------------------------------

typedef __attribute__((ext_vector_type(16))) _Float16 v16h;
typedef __attribute__((ext_vector_type(8)))  _Float16 v8h;
typedef __attribute__((ext_vector_type(8)))  float    v8f;
typedef __attribute__((ext_vector_type(4)))  unsigned int u32x4;
typedef __attribute__((ext_vector_type(8)))  int          i32x8;
typedef __attribute__((ext_vector_type(4)))  int          i32x4;

#define DIMX   512
#define HEADS  16
#define DHEAD  64
#define INNER  1024
#define BATCH  4
#define TSEQ   2048
#define MTOT   (BATCH * TSEQ)   // 8192

#if __has_builtin(__builtin_amdgcn_tensor_load_to_lds) && \
    __has_builtin(__builtin_amdgcn_s_wait_tensorcnt)
#define USE_TDM 1
#else
#define USE_TDM 0
#endif

static __device__ __forceinline__ v8f wmma16(v16h a, v16h b, v8f c) {
    // D = A(16x32 f16) * B(32x16 f16) + C(16x16 f32)
    return __builtin_amdgcn_wmma_f32_16x16x32_f16(
        /*neg_a=*/false, a, /*neg_b=*/false, b,
        /*c_mod=*/(short)0, c, /*reuse_a=*/false, /*reuse_b=*/false);
}

// A-fragment (16x32, MxK), contiguous k order (same as load_bfrag):
//  lane L   (<16): row m0+L, K = k0+0..15
//  lane L+16     : row m0+L, K = k0+16..31
static __device__ __forceinline__ v16h load_afrag(const _Float16* base, int lda,
                                                  int m0, int k0) {
    const int lane = threadIdx.x & 31;
    const int mrow = lane & 15;
    const int half = lane >> 4;
    const _Float16* p = base + (size_t)(m0 + mrow) * lda + k0 + half * 16;
    v8h lo = *(const v8h*)p;          // K: k0+half*16   .. +7
    v8h hi = *(const v8h*)(p + 8);    // K: k0+half*16+8 .. +7
    return __builtin_shufflevector(lo, hi, 0, 1, 2, 3, 4, 5, 6, 7,
                                           8, 9, 10, 11, 12, 13, 14, 15);
}

// B-fragment (32x16, KxN) read from an N-major ("Bt", N x K row-major) array:
//  lane L   (<16): column n0+L, K = k0+0..15   (contiguous 32B)
//  lane L+16     : column n0+L, K = k0+16..31
static __device__ __forceinline__ v16h load_bfrag(const _Float16* bt, int ldb,
                                                  int n0, int k0) {
    const int lane = threadIdx.x & 31;
    const int mrow = lane & 15;
    const int half = lane >> 4;
    return *(const v16h*)(bt + (size_t)(n0 + mrow) * ldb + k0 + half * 16);
}

#if USE_TDM
// Issue one TDM descriptor: 32 rows x 64 f16 (row stride 64) -> LDS, 2D tile.
// D# layout per CDNA5 ISA.
static __device__ __forceinline__ void tdm_load_tile_32x64(const _Float16* gsrc,
                                                           unsigned lds_off) {
    const unsigned long long ga = (unsigned long long)gsrc;
    u32x4 g0;
    g0[0] = 1u;                                    // count=1 (valid), user mode
    g0[1] = lds_off;                               // D#.lds_addr (bytes)
    g0[2] = (unsigned)(ga & 0xFFFFFFFFu);          // D#.global_addr[31:0]
    g0[3] = (unsigned)((ga >> 32) & 0x01FFFFFFu)   // D#.global_addr[56:32]
          | (2u << 30);                            // D#.type = 2
    i32x8 g1;
    g1[0] = 1 << 16;        // data_size=1 (2 bytes/elem), workgroup_mask=0
    g1[1] = 64 << 16;       // tensor_dim0[15:0] = 64 elems per row
    g1[2] = 2048 << 16;     // tensor_dim0 hi=0 ; tensor_dim1[15:0] = 2048 rows
    g1[3] = 64 << 16;       // tensor_dim1 hi=0 ; tile_dim0 = 64
    g1[4] = 32;             // tile_dim1 = 32 rows ; tile_dim2 = 0
    g1[5] = 64;             // tensor_dim0_stride[31:0] = 64 elems
    g1[6] = 0;              // stride hi / dim1_stride lo (unused, 2D tile)
    g1[7] = 0;
    const i32x4 z4 = {0, 0, 0, 0};
#if defined(__clang_major__) && (__clang_major__ >= 23)
    const i32x8 z8 = {0, 0, 0, 0, 0, 0, 0, 0};
    __builtin_amdgcn_tensor_load_to_lds(g0, g1, z4, z4, z8, 0);
#else
    __builtin_amdgcn_tensor_load_to_lds(g0, g1, z4, z4, 0);
#endif
}
#endif

// ---------------------------------------------------------------------------
// Precision conversion kernels
// ---------------------------------------------------------------------------
__global__ void k_cvt(const float* __restrict__ in, _Float16* __restrict__ out, int n) {
    int i = blockIdx.x * blockDim.x + threadIdx.x;
    if (i < n) out[i] = (_Float16)in[i];
}

// in: R x C row-major f32 -> out: C x R row-major f16 (transposed)
__global__ void k_cvt_t(const float* __restrict__ in, _Float16* __restrict__ out,
                        int R, int C) {
    int i = blockIdx.x * blockDim.x + threadIdx.x;
    if (i < R * C) {
        int r = i / C, c = i - r * C;
        out[(size_t)c * R + r] = (_Float16)in[i];
    }
}

// ---------------------------------------------------------------------------
// QKV projection: [8192 x 512] @ [512 x 1024] per weight, results stored as
// [b, h, t, d] f16. 4 waves/block; wave tile = 32x64 (8 WMMA / K-step),
// software-pipelined fragment prefetch, unroll-4 to kill rotation copies.
// ---------------------------------------------------------------------------
__global__ __launch_bounds__(128) void k_qkv(
        const _Float16* __restrict__ X,
        const _Float16* __restrict__ Wqt, const _Float16* __restrict__ Wkt,
        const _Float16* __restrict__ Wvt,
        _Float16* __restrict__ Q, _Float16* __restrict__ K,
        _Float16* __restrict__ V) {
    const int w    = threadIdx.x >> 5;
    const int lane = threadIdx.x & 31;
    const int mrow = lane & 15, half = lane >> 4;
    const int m0 = blockIdx.x * 128 + w * 32;
    const int n0 = blockIdx.y * 64;
    const _Float16* Wt  = (blockIdx.z == 0) ? Wqt : (blockIdx.z == 1) ? Wkt : Wvt;
    _Float16*       Dst = (blockIdx.z == 0) ? Q   : (blockIdx.z == 1) ? K   : V;

    v8f acc[8] = {v8f{}, v8f{}, v8f{}, v8f{}, v8f{}, v8f{}, v8f{}, v8f{}};

    v16h a0 = load_afrag(X, DIMX, m0, 0);
    v16h a1 = load_afrag(X, DIMX, m0 + 16, 0);
    v16h bb[4];
#pragma unroll
    for (int j = 0; j < 4; ++j) bb[j] = load_bfrag(Wt, DIMX, n0 + j * 16, 0);

#pragma unroll 4
    for (int kk = 0; kk < DIMX; kk += 32) {
        const int kn = (kk + 32 < DIMX) ? kk + 32 : kk;  // clamped prefetch
        v16h a0n = load_afrag(X, DIMX, m0, kn);
        v16h a1n = load_afrag(X, DIMX, m0 + 16, kn);
        v16h bn[4];
#pragma unroll
        for (int j = 0; j < 4; ++j) bn[j] = load_bfrag(Wt, DIMX, n0 + j * 16, kn);
#pragma unroll
        for (int j = 0; j < 4; ++j) acc[j] = wmma16(a0, bb[j], acc[j]);
#pragma unroll
        for (int j = 0; j < 4; ++j) acc[4 + j] = wmma16(a1, bb[j], acc[4 + j]);
        a0 = a0n; a1 = a1n;
#pragma unroll
        for (int j = 0; j < 4; ++j) bb[j] = bn[j];
    }

#pragma unroll
    for (int part = 0; part < 2; ++part) {
#pragma unroll
        for (int j = 0; j < 4; ++j) {
            const int n = n0 + j * 16 + mrow;
            const int h = n >> 6, d = n & 63;
#pragma unroll
            for (int r = 0; r < 8; ++r) {
                const int m  = m0 + part * 16 + half * 8 + r;
                const int bb2 = m >> 11, t = m & 2047;
                Dst[((size_t)((bb2 * HEADS + h) * TSEQ + t)) * DHEAD + d] =
                    (_Float16)acc[part * 4 + j][r];
            }
        }
    }
}

// ---------------------------------------------------------------------------
// Flash attention: block = 4 waves = 64 query rows of one (b,h); wave = 16 rows.
// 32-key tiles staged by the Tensor Data Mover (double-buffered), S = Q*K^T
// (4 WMMA), online softmax, O += P*V (4 WMMA). Unroll-2 makes the buffer
// index compile-time constant per copy.
// ---------------------------------------------------------------------------
__global__ __launch_bounds__(128) void k_attn(
        const _Float16* __restrict__ Qg, const _Float16* __restrict__ Kg,
        const _Float16* __restrict__ Vg, _Float16* __restrict__ AO) {
#if USE_TDM
    __shared__ _Float16 Kraw[2][32][64];   // K tile, row-major (TDM target)
    __shared__ _Float16 Vraw[2][32][64];   // V tile, row-major (TDM target)
#endif
    __shared__ _Float16 Vt[2][64][32];     // V tile transposed: Vt[d][k]
    __shared__ _Float16 Pw[4][16][32];     // per-wave P tile (f16)

    const int w    = threadIdx.x >> 5;
    const int lane = threadIdx.x & 31;
    const int mrow = lane & 15, half = lane >> 4;
    const int bh = blockIdx.z * HEADS + blockIdx.y;
    const _Float16* Qb = Qg + (size_t)bh * TSEQ * DHEAD;
    const _Float16* Kb = Kg + (size_t)bh * TSEQ * DHEAD;
    const _Float16* Vb = Vg + (size_t)bh * TSEQ * DHEAD;
    const int t0 = blockIdx.x * 64 + w * 16;

    // Q fragments for this wave's 16 rows (K = 64 split into two 32-chunks)
    const v16h aq0 = load_afrag(Qb, DHEAD, t0, 0);
    const v16h aq1 = load_afrag(Qb, DHEAD, t0, 32);

    float m_r[8], l_r[8];
#pragma unroll
    for (int r = 0; r < 8; ++r) { m_r[r] = -3.0e38f; l_r[r] = 0.0f; }
    v8f acc[4] = {v8f{}, v8f{}, v8f{}, v8f{}};
    const float scale = 0.125f;  // 1/sqrt(64)

#if USE_TDM
    if (w == 0) {   // prologue: DMA first K/V tiles into buffer 0
        tdm_load_tile_32x64(Kb, (unsigned)(unsigned long long)&Kraw[0][0][0]);
        tdm_load_tile_32x64(Vb, (unsigned)(unsigned long long)&Vraw[0][0][0]);
    }
#endif

#pragma unroll 2
    for (int kt = 0; kt < TSEQ; kt += 32) {
        const int b = (kt >> 5) & 1;

#if USE_TDM
        __syncthreads();   // everyone done reading buffer b^1 (iter kt-32)
        if (w == 0) {
            if (kt + 32 < TSEQ) {   // DMA next tiles into the free buffer
                tdm_load_tile_32x64(Kb + (size_t)(kt + 32) * DHEAD,
                                    (unsigned)(unsigned long long)&Kraw[b ^ 1][0][0]);
                tdm_load_tile_32x64(Vb + (size_t)(kt + 32) * DHEAD,
                                    (unsigned)(unsigned long long)&Vraw[b ^ 1][0][0]);
                __builtin_amdgcn_s_wait_tensorcnt(2);  // current tiles done
            } else {
                __builtin_amdgcn_s_wait_tensorcnt(0);
            }
        }
        __syncthreads();   // Kraw[b]/Vraw[b] visible to all waves
        {   // LDS->LDS transpose of V tile: Vraw[b] -> Vt[b]
            const int k    = threadIdx.x >> 2;
            const int dseg = (threadIdx.x & 3) * 16;
            v16h vv = *(const v16h*)(&Vraw[b][k][dseg]);
#pragma unroll
            for (int j = 0; j < 16; ++j) Vt[b][dseg + j][k] = vv[j];
        }
        __syncthreads();   // Vt[b] ready
        const _Float16* Ktile = &Kraw[b][0][0];
        const int kbase = 0, kld = 64;
#else
        {   // fallback: stage V tile transposed straight from global
            const int k    = threadIdx.x >> 2;
            const int dseg = (threadIdx.x & 3) * 16;
            v16h vv = *(const v16h*)(Vb + (size_t)(kt + k) * DHEAD + dseg);
#pragma unroll
            for (int j = 0; j < 16; ++j) Vt[b][dseg + j][k] = vv[j];
        }
        __syncthreads();   // Vt[b] ready (double-buffered: one barrier/iter)
        if (kt + 32 < TSEQ) {   // global_prefetch_b8 next K/V tiles
            __builtin_prefetch(Kb + (size_t)(kt + 32) * DHEAD, 0, 0);
            __builtin_prefetch(Vb + (size_t)(kt + 32) * DHEAD, 0, 0);
        }
        const _Float16* Ktile = Kb;
        const int kbase = kt, kld = DHEAD;
#endif

        // S tile: 16 queries x 32 keys. B = K^T: columns of B are rows of K.
        v8f s0 = {}, s1 = {};
        {
            v16h b00 = load_bfrag(Ktile, kld, kbase + 0,  0);
            v16h b01 = load_bfrag(Ktile, kld, kbase + 0,  32);
            v16h b10 = load_bfrag(Ktile, kld, kbase + 16, 0);
            v16h b11 = load_bfrag(Ktile, kld, kbase + 16, 32);
            s0 = wmma16(aq0, b00, s0);
            s0 = wmma16(aq1, b01, s0);
            s1 = wmma16(aq0, b10, s1);
            s1 = wmma16(aq1, b11, s1);
        }

        // Online softmax: row r lives in v[r] across 16 lanes of a half-wave
        // (lanes<16: rows 0-7, lanes>=16: rows 8-15).
#pragma unroll
        for (int r = 0; r < 8; ++r) {
            float a0 = s0[r] * scale, a1 = s1[r] * scale;
            float mx = fmaxf(a0, a1);
            mx = fmaxf(mx, __shfl_xor(mx, 1, 32));
            mx = fmaxf(mx, __shfl_xor(mx, 2, 32));
            mx = fmaxf(mx, __shfl_xor(mx, 4, 32));
            mx = fmaxf(mx, __shfl_xor(mx, 8, 32));
            const float nm    = fmaxf(m_r[r], mx);
            const float alpha = __expf(m_r[r] - nm);
            m_r[r] = nm;
            a0 = __expf(a0 - nm);
            a1 = __expf(a1 - nm);
            float rs = a0 + a1;
            rs += __shfl_xor(rs, 1, 32);
            rs += __shfl_xor(rs, 2, 32);
            rs += __shfl_xor(rs, 4, 32);
            rs += __shfl_xor(rs, 8, 32);
            l_r[r] = l_r[r] * alpha + rs;
#pragma unroll
            for (int j = 0; j < 4; ++j) acc[j][r] = acc[j][r] * alpha;
            // spill P (f16) to LDS to re-fragment D-layout -> A-layout
            Pw[w][half * 8 + r][mrow]      = (_Float16)a0;
            Pw[w][half * 8 + r][mrow + 16] = (_Float16)a1;
        }
        __builtin_amdgcn_wave_barrier();  // LDS is in-order per wave; fence
                                          // compiler scheduling only.
        {   // O += P(16x32) * V(32x64)
            const _Float16* pb = &Pw[w][0][0];
            v8h lo = *(const v8h*)(pb + mrow * 32 + half * 16);
            v8h hi = *(const v8h*)(pb + mrow * 32 + half * 16 + 8);
            v16h ap = __builtin_shufflevector(lo, hi, 0, 1, 2, 3, 4, 5, 6, 7,
                                              8, 9, 10, 11, 12, 13, 14, 15);
#pragma unroll
            for (int j = 0; j < 4; ++j) {
                v16h bv = *(const v16h*)(&Vt[b][j * 16 + mrow][half * 16]);
                acc[j] = wmma16(ap, bv, acc[j]);
            }
        }
    }

    // Epilogue: O /= l, store to [b, t, h*64+d] f16 for the output projection.
    const int bb = blockIdx.z, h = blockIdx.y;
#pragma unroll
    for (int r = 0; r < 8; ++r) {
        const float inv = 1.0f / l_r[r];
        const int   t   = t0 + half * 8 + r;
#pragma unroll
        for (int j = 0; j < 4; ++j) {
            const int d = j * 16 + mrow;
            AO[((size_t)(bb * TSEQ + t)) * INNER + h * DHEAD + d] =
                (_Float16)(acc[j][r] * inv);
        }
    }
}

// ---------------------------------------------------------------------------
// Output projection: [8192 x 1024] @ [1024 x 512] + bias, f32 out.
// Wave tile = 32x64, software-pipelined fragment prefetch, unroll-4.
// ---------------------------------------------------------------------------
__global__ __launch_bounds__(128) void k_out(
        const _Float16* __restrict__ AO, const _Float16* __restrict__ Wot,
        const float* __restrict__ bo, float* __restrict__ Out) {
    const int w    = threadIdx.x >> 5;
    const int lane = threadIdx.x & 31;
    const int mrow = lane & 15, half = lane >> 4;
    const int m0 = blockIdx.x * 128 + w * 32;
    const int n0 = blockIdx.y * 64;

    v8f acc[8] = {v8f{}, v8f{}, v8f{}, v8f{}, v8f{}, v8f{}, v8f{}, v8f{}};

    v16h a0 = load_afrag(AO, INNER, m0, 0);
    v16h a1 = load_afrag(AO, INNER, m0 + 16, 0);
    v16h bb[4];
#pragma unroll
    for (int j = 0; j < 4; ++j) bb[j] = load_bfrag(Wot, INNER, n0 + j * 16, 0);

#pragma unroll 4
    for (int kk = 0; kk < INNER; kk += 32) {
        const int kn = (kk + 32 < INNER) ? kk + 32 : kk;  // clamped prefetch
        v16h a0n = load_afrag(AO, INNER, m0, kn);
        v16h a1n = load_afrag(AO, INNER, m0 + 16, kn);
        v16h bn[4];
#pragma unroll
        for (int j = 0; j < 4; ++j) bn[j] = load_bfrag(Wot, INNER, n0 + j * 16, kn);
#pragma unroll
        for (int j = 0; j < 4; ++j) acc[j] = wmma16(a0, bb[j], acc[j]);
#pragma unroll
        for (int j = 0; j < 4; ++j) acc[4 + j] = wmma16(a1, bb[j], acc[4 + j]);
        a0 = a0n; a1 = a1n;
#pragma unroll
        for (int j = 0; j < 4; ++j) bb[j] = bn[j];
    }

#pragma unroll
    for (int part = 0; part < 2; ++part) {
#pragma unroll
        for (int j = 0; j < 4; ++j) {
            const int   n    = n0 + j * 16 + mrow;
            const float bias = bo[n];
#pragma unroll
            for (int r = 0; r < 8; ++r) {
                const int m = m0 + part * 16 + half * 8 + r;
                Out[(size_t)m * DIMX + n] = acc[part * 4 + j][r] + bias;
            }
        }
    }
}

// ---------------------------------------------------------------------------
extern "C" void kernel_launch(void* const* d_in, const int* in_sizes, int n_in,
                              void* d_out, int out_size, void* d_ws, size_t ws_size,
                              hipStream_t stream) {
    const float* x  = (const float*)d_in[0];
    const float* Wq = (const float*)d_in[1];
    const float* Wk = (const float*)d_in[2];
    const float* Wv = (const float*)d_in[3];
    const float* Wo = (const float*)d_in[4];
    const float* bo = (const float*)d_in[5];

    char* ws = (char*)d_ws;
    size_t off = 0;
    _Float16* xh  = (_Float16*)(ws + off); off += (size_t)MTOT * DIMX * 2;
    _Float16* Wqt = (_Float16*)(ws + off); off += (size_t)INNER * DIMX * 2;
    _Float16* Wkt = (_Float16*)(ws + off); off += (size_t)INNER * DIMX * 2;
    _Float16* Wvt = (_Float16*)(ws + off); off += (size_t)INNER * DIMX * 2;
    _Float16* Wot = (_Float16*)(ws + off); off += (size_t)DIMX * INNER * 2;
    _Float16* Qh  = (_Float16*)(ws + off); off += (size_t)MTOT * INNER * 2;
    _Float16* Kh  = (_Float16*)(ws + off); off += (size_t)MTOT * INNER * 2;
    _Float16* Vh  = (_Float16*)(ws + off); off += (size_t)MTOT * INNER * 2;
    _Float16* AO  = (_Float16*)(ws + off); off += (size_t)MTOT * INNER * 2;

    // 1) precision conversion / weight transposition
    k_cvt<<<(MTOT * DIMX + 255) / 256, 256, 0, stream>>>(x, xh, MTOT * DIMX);
    k_cvt_t<<<(DIMX * INNER + 255) / 256, 256, 0, stream>>>(Wq, Wqt, DIMX, INNER);
    k_cvt_t<<<(DIMX * INNER + 255) / 256, 256, 0, stream>>>(Wk, Wkt, DIMX, INNER);
    k_cvt_t<<<(DIMX * INNER + 255) / 256, 256, 0, stream>>>(Wv, Wvt, DIMX, INNER);
    k_cvt_t<<<(INNER * DIMX + 255) / 256, 256, 0, stream>>>(Wo, Wot, INNER, DIMX);

    // 2) QKV projection (WMMA, pipelined)
    k_qkv<<<dim3(MTOT / 128, INNER / 64, 3), 128, 0, stream>>>(
        xh, Wqt, Wkt, Wvt, Qh, Kh, Vh);

    // 3) flash attention (WMMA + TDM staging + online softmax)
    k_attn<<<dim3(TSEQ / 64, HEADS, BATCH), 128, 0, stream>>>(Qh, Kh, Vh, AO);

    // 4) output projection + bias (WMMA, f32 out)
    k_out<<<dim3(MTOT / 128, DIMX / 64), 128, 0, stream>>>(
        AO, Wot, bo, (float*)d_out);
}
